// LSTM_4784593568414
// MI455X (gfx1250) — matrix operand order repaired and checked by reading, not compiled, not run on hardware
//
#include <hip/hip_runtime.h>
#include <hip/hip_bf16.h>

typedef __attribute__((ext_vector_type(16))) __bf16 v16bf;
typedef __attribute__((ext_vector_type(8)))  __bf16 v8bf;
typedef __attribute__((ext_vector_type(8)))  float  v8f;
typedef int v4i_ __attribute__((vector_size(16)));          // matches builtin proto
typedef __attribute__((address_space(1))) v4i_ g_v4i;       // global (AS1)
typedef __attribute__((address_space(3))) v4i_ l_v4i;       // LDS (AS3)

#define Bx 32
#define Sx 512
#define Ix 512
#define Hx 1024
#define Ox 512
#define G4 4096
#define FINAL_ELEMS (Bx*Sx*Ox)   // 8388608 floats in d_out before (h_fin, c_fin)
#define NWG_SCAN 64

#if __has_builtin(__builtin_amdgcn_global_load_async_to_lds_b128)
#define ASYNC_LDS 1
#endif

// ---------- workspace layout (bytes) ----------
#define OFF_XBF   ((size_t)0)                       // B*S*I bf16      = 16 MB
#define OFF_WX    (OFF_XBF  + (size_t)Bx*Sx*Ix*2)   // 4096*512 bf16   =  4 MB
#define OFF_WH    (OFF_WX   + (size_t)G4*Ix*2)      // 4096*1024 bf16  =  8 MB
#define OFF_WO    (OFF_WH   + (size_t)G4*Hx*2)      // 512*1024 bf16   =  1 MB
#define OFF_B4    (OFF_WO   + (size_t)Ox*Hx*2)      // 4096 f32
#define OFF_XG    (OFF_B4   + (size_t)G4*4)         // S*4096*32 bf16  = 128 MB
#define OFF_HS    (OFF_XG   + (size_t)Sx*G4*Bx*2)   // B*S*H bf16      = 32 MB
#define OFF_HCUR  (OFF_HS   + (size_t)Bx*Sx*Hx*2)   // B*H bf16
#define OFF_SYNC  (OFF_HCUR + (size_t)Bx*Hx*2)      // 256 B

// ---------- WMMA helpers (CDNA5 bf16 16x16x32, wave32) ----------
__device__ __forceinline__ v16bf afrag(const __bf16* p) {
  // A 16x32 bf16: lane L holds row M=L&15; elems 0..7 -> K = 8*(L>>4)+e,
  // elems 8..15 -> K = 16 + 8*(L>>4) + (e-8). Two contiguous 16B loads.
  union { v16bf v; v8bf h[2]; } u;
  u.h[0] = *(const v8bf*)(p);
  u.h[1] = *(const v8bf*)(p + 16);
  return u.v;
}
__device__ __forceinline__ v8f wmma_bf16(v16bf a, v16bf b, v8f c) {
  return __builtin_amdgcn_wmma_f32_16x16x32_bf16(false, a, false, b, (short)0, c,
                                                 false, false);
}
__device__ __forceinline__ float sigm(float x) { return 1.0f / (1.0f + __expf(-x)); }

// ---------- prep kernels ----------
__global__ __launch_bounds__(256) void cvt_x_kernel(const float* __restrict__ x,
                                                    __bf16* __restrict__ xb) {
  size_t i = (size_t)blockIdx.x * blockDim.x + threadIdx.x;
  if (i < (size_t)Bx*Sx*Ix) xb[i] = (__bf16)x[i];
}

__global__ __launch_bounds__(256) void pack_gates_kernel(
    const float* __restrict__ Wf, const float* __restrict__ Wi,
    const float* __restrict__ Wo, const float* __restrict__ Wc,
    __bf16* __restrict__ Wx, __bf16* __restrict__ Wh) {
  size_t idx = (size_t)blockIdx.x * blockDim.x + threadIdx.x;
  if (idx >= (size_t)4*1024*(Ix+Hx)) return;
  int c  = (int)(idx % (Ix+Hx));
  int r  = (int)((idx / (Ix+Hx)) % 1024);
  int gb = (int)(idx / ((size_t)(Ix+Hx)*1024));
  const float* src = (gb==0) ? Wf : (gb==1) ? Wi : (gb==2) ? Wo : Wc;
  float v = src[(size_t)r*(Ix+Hx) + c];
  int g = gb*1024 + r;
  if (c < Ix) Wx[(size_t)g*Ix + c] = (__bf16)v;
  else        Wh[(size_t)g*Hx + (c - Ix)] = (__bf16)v;
}

__global__ __launch_bounds__(256) void pack_bias_kernel(
    const float* __restrict__ bf, const float* __restrict__ bi,
    const float* __restrict__ bo, const float* __restrict__ bc,
    float* __restrict__ b4) {
  int idx = blockIdx.x * blockDim.x + threadIdx.x;
  if (idx >= G4) return;
  int gb = idx >> 10, r = idx & 1023;
  const float* src = (gb==0) ? bf : (gb==1) ? bi : (gb==2) ? bo : bc;
  b4[idx] = src[r];
}

__global__ __launch_bounds__(256) void pack_outw_kernel(const float* __restrict__ ow,
                                                        __bf16* __restrict__ owb) {
  size_t i = (size_t)blockIdx.x * blockDim.x + threadIdx.x;
  if (i < (size_t)Ox*Hx) owb[i] = (__bf16)ow[i];
}

// ---------- xg = x @ Wx^T + b : D[s][g][b], bf16 out ----------
// tile: M=g(16), N=b(16), K=i(512). 262144 tiles, 8 waves/block.
__global__ __launch_bounds__(256) void xg_gemm_kernel(
    const __bf16* __restrict__ xb, const __bf16* __restrict__ Wx,
    const float* __restrict__ b4, __bf16* __restrict__ xg) {
  int tid = threadIdx.x, wave = tid >> 5, lane = tid & 31;
  int lh = lane >> 4, ln = lane & 15;
  int tile = blockIdx.x * 8 + wave;
  int s = tile >> 9, rem = tile & 511;
  int g0 = (rem >> 1) * 16, b0 = (rem & 1) * 16;

  v8f acc;
  const float* bp = b4 + g0 + 8*lh;          // C: M=g = g0 + r + 8*lh
  #pragma unroll
  for (int r = 0; r < 8; ++r) acc[r] = bp[r];

  const __bf16* arow = Wx + (size_t)(g0 + ln) * Ix;                 // A row = gate row
  const __bf16* brow = xb + ((size_t)(b0 + ln) * Sx + s) * Ix;     // B col = batch

  #pragma unroll 4
  for (int ks = 0; ks < Ix/32; ++ks) {
    v16bf a  = afrag(arow + 32*ks + 8*lh);
    v16bf bb = afrag(brow + 32*ks + 8*lh);
    acc = wmma_bf16(a, bb, acc);
  }
  #pragma unroll
  for (int r = 0; r < 8; ++r) {
    size_t o = ((size_t)s*G4 + g0 + r + 8*lh) * Bx + b0 + ln;
    xg[o] = (__bf16)acc[r];
  }
}

// ---------- persistent recurrent scan: 64 WGs, Wh slice LDS-resident ----------
__global__ __launch_bounds__(256, 1) void scan_kernel(
    const __bf16* __restrict__ xg, const __bf16* __restrict__ Wh,
    __bf16* __restrict__ hcur, __bf16* __restrict__ hs,
    float* __restrict__ out_tail, int* __restrict__ sync) {
  extern __shared__ char smem[];
  __bf16* whs = (__bf16*)smem;                    // [64][1024] bf16 = 128 KB
  float*  gex = (float*)(smem + 64*1024*2);       // [4][32][16] f32 = 8 KB
  float*  cst = (float*)(smem + 64*1024*2 + 8192);// [32][16] f32    = 2 KB

  int tid = threadIdx.x, wave = tid >> 5, lane = tid & 31;
  int lh = lane >> 4, ln = lane & 15;
  int wg = blockIdx.x, j0 = wg * 16;              // hidden units [j0, j0+16)
  int gb = wave >> 1, mt = wave & 1;              // gate block, M half (batches)

  // stage this WG's 64 gate rows of Wh into LDS (once, reused 512 steps)
#ifdef ASYNC_LDS
  for (int idx = tid; idx < 64*128; idx += 256) {  // 16B chunks, ASYNCcnt path
    int row = idx >> 7, c8 = idx & 127;
    int grow = (row >> 4) * Hx + j0 + (row & 15);
    __builtin_amdgcn_global_load_async_to_lds_b128(
        (g_v4i*)(Wh + (size_t)grow*Hx + c8*8),
        (l_v4i*)(whs + row*Hx + c8*8), 0, 0);
  }
  asm volatile("s_wait_asynccnt 0" ::: "memory");
#else
  for (int idx = tid; idx < 64*128; idx += 256) {  // 16B chunks
    int row = idx >> 7, c8 = idx & 127;
    int grow = (row >> 4) * Hx + j0 + (row & 15);
    *(uint4*)(whs + row*Hx + c8*8) =
        *(const uint4*)(Wh + (size_t)grow*Hx + c8*8);
  }
#endif
  for (int idx = tid; idx < Bx*16; idx += 256) cst[idx] = 0.0f;
  __syncthreads();

  for (int s = 0; s < Sx; ++s) {
    // --- C-init from precomputed xg (already includes bias); layout [s][g][b]
    const __bf16* xgl = xg + ((size_t)s*G4 + gb*Hx + j0 + ln) * Bx + mt*16 + 8*lh;
    if (s + 1 < Sx)
      __builtin_prefetch(xgl + (size_t)G4*Bx, 0, 0);
    v8bf xv = *(const v8bf*)xgl;
    v8f acc;
    #pragma unroll
    for (int r = 0; r < 8; ++r) acc[r] = (float)xv[r];

    // --- acc += h @ Wh^T  (A = h from global, B = Wh slice from LDS)
    if (s > 0) {
      const __bf16* hrow = hcur + (size_t)(mt*16 + ln) * Hx;   // A row = batch
      const __bf16* wrow = whs + (gb*16 + ln) * Hx;            // B col = gate row
      #pragma unroll 4
      for (int ks = 0; ks < Hx/32; ++ks) {
        v16bf a  = afrag(hrow + 32*ks + 8*lh);
        v16bf bb = afrag(wrow + 32*ks + 8*lh);
        acc = wmma_bf16(a, bb, acc);
      }
    }
    // --- exchange gate tiles through LDS: gex[gb][b][j]
    #pragma unroll
    for (int r = 0; r < 8; ++r)
      gex[(gb*Bx + mt*16 + r + 8*lh) * 16 + ln] = acc[r];
    __syncthreads();

    // --- fused cell update: 512 (b,j) pairs over 256 threads
    #pragma unroll
    for (int q = 0; q < 2; ++q) {
      int idx = tid + q*256;
      int b = idx >> 4, j = idx & 15;
      float f  = sigm(gex[(0*Bx + b)*16 + j]);
      float ig = sigm(gex[(1*Bx + b)*16 + j]);
      float og = sigm(gex[(2*Bx + b)*16 + j]);
      float cc = tanhf(gex[(3*Bx + b)*16 + j]);
      float c = f * cst[idx] + ig * cc;
      cst[idx] = c;
      float h = og * tanhf(c);
      __bf16 hb = (__bf16)h;
      hcur[(size_t)b*Hx + j0 + j] = hb;
      hs[((size_t)b*Sx + s)*Hx + j0 + j] = hb;
      if (s == Sx-1) {
        out_tail[FINAL_ELEMS + (size_t)b*Hx + j0 + j] = h;           // h_fin
        out_tail[FINAL_ELEMS + (size_t)Bx*Hx + (size_t)b*Hx + j0 + j] = c; // c_fin
      }
    }
    __threadfence();
    __syncthreads();

    // --- grid-wide step barrier (monotonic counter, release/acquire)
    if (s + 1 < Sx) {
      if (tid == 0) {
        __hip_atomic_fetch_add(sync, 1, __ATOMIC_RELEASE, __HIP_MEMORY_SCOPE_AGENT);
        int target = NWG_SCAN * (s + 1);
        while (__hip_atomic_load(sync, __ATOMIC_ACQUIRE, __HIP_MEMORY_SCOPE_AGENT) < target)
          __builtin_amdgcn_s_sleep(2);
      }
      __syncthreads();
    }
  }
}

// ---------- final = hs @ out_w^T + out_b : fp32 out ----------
// M=(b,s) rows (1024 tiles), N=o (32 tiles), K=H=1024.
__global__ __launch_bounds__(256) void out_proj_kernel(
    const __bf16* __restrict__ hs, const __bf16* __restrict__ Wo,
    const float* __restrict__ ob, float* __restrict__ out) {
  int tid = threadIdx.x, wave = tid >> 5, lane = tid & 31;
  int lh = lane >> 4, ln = lane & 15;
  int tile = blockIdx.x * 8 + wave;
  int r0 = (tile >> 5) * 16, o0 = (tile & 31) * 16;

  float bias = ob[o0 + ln];
  v8f acc;
  #pragma unroll
  for (int r = 0; r < 8; ++r) acc[r] = bias;

  const __bf16* arow = hs + (size_t)(r0 + ln) * Hx;
  const __bf16* brow = Wo + (size_t)(o0 + ln) * Hx;
  #pragma unroll 4
  for (int ks = 0; ks < Hx/32; ++ks) {
    v16bf a  = afrag(arow + 32*ks + 8*lh);
    v16bf bb = afrag(brow + 32*ks + 8*lh);
    acc = wmma_bf16(a, bb, acc);
  }
  #pragma unroll
  for (int r = 0; r < 8; ++r)
    out[(size_t)(r0 + r + 8*lh) * Ox + o0 + ln] = acc[r];
}

extern "C" void kernel_launch(void* const* d_in, const int* in_sizes, int n_in,
                              void* d_out, int out_size, void* d_ws, size_t ws_size,
                              hipStream_t stream) {
  (void)in_sizes; (void)n_in; (void)out_size; (void)ws_size;
  const float* x    = (const float*)d_in[0];
  const float* Wf_w = (const float*)d_in[1];
  const float* Wf_b = (const float*)d_in[2];
  const float* Wi_w = (const float*)d_in[3];
  const float* Wi_b = (const float*)d_in[4];
  const float* Wo_w = (const float*)d_in[5];
  const float* Wo_b = (const float*)d_in[6];
  const float* Wc_w = (const float*)d_in[7];
  const float* Wc_b = (const float*)d_in[8];
  const float* ow   = (const float*)d_in[9];
  const float* obias= (const float*)d_in[10];

  char* ws = (char*)d_ws;
  __bf16* xb   = (__bf16*)(ws + OFF_XBF);
  __bf16* Wx   = (__bf16*)(ws + OFF_WX);
  __bf16* Wh   = (__bf16*)(ws + OFF_WH);
  __bf16* Wob  = (__bf16*)(ws + OFF_WO);
  float*  b4   = (float*) (ws + OFF_B4);
  __bf16* xg   = (__bf16*)(ws + OFF_XG);
  __bf16* hs   = (__bf16*)(ws + OFF_HS);
  __bf16* hcur = (__bf16*)(ws + OFF_HCUR);
  int*    sync = (int*)   (ws + OFF_SYNC);
  float*  out  = (float*)d_out;

  (void)hipMemsetAsync(sync, 0, 256, stream);   // deterministic barrier counter

  cvt_x_kernel<<<(Bx*Sx*Ix)/256, 256, 0, stream>>>(x, xb);
  pack_gates_kernel<<<(4*1024*(Ix+Hx))/256, 256, 0, stream>>>(Wf_w, Wi_w, Wo_w, Wc_w, Wx, Wh);
  pack_bias_kernel<<<G4/256, 256, 0, stream>>>(Wf_b, Wi_b, Wo_b, Wc_b, b4);
  pack_outw_kernel<<<(Ox*Hx)/256, 256, 0, stream>>>(ow, Wob);

  // 512 s * 256 gtiles * 2 btiles = 262144 tiles, 8 waves/block
  xg_gemm_kernel<<<262144/8, 256, 0, stream>>>(xb, Wx, b4, xg);

  size_t smem = (size_t)64*1024*2 + 8192 + 2048;   // 141312 B < 320 KB WGP LDS
  scan_kernel<<<NWG_SCAN, 256, smem, stream>>>(xg, Wh, hcur, hs, out, sync);

  // 1024 mtiles * 32 ntiles = 32768 tiles, 8 waves/block
  out_proj_kernel<<<32768/8, 256, 0, stream>>>(hs, Wob, obias, out);
}
